// NequIP_39024072851680
// MI455X (gfx1250) — matrix operand order repaired and checked by reading, not compiled, not run on hardware
//
#include <hip/hip_runtime.h>

#define N_NODES  20000
#define N_EDGES  320000
#define N_GRAPHS 50

typedef __attribute__((ext_vector_type(16))) _Float16 v16h;
typedef __attribute__((ext_vector_type(8)))  float    v8f;

#define PI_F        3.14159265358979323846f
#define SQRT3_F     1.7320508075688772f
#define INV_SQRT3_F 0.5773502691896258f
#define C_S_F       0.3826834323650898f   /* sin(pi/8) */
#define C_X_F       0.9238795325112867f   /* cos(pi/8) */
#define INV_SQRT8   0.35355339059327373f
#define INV_SQRT32  0.17677669529663687f
#define INV_SQRT96  0.10206207261596577f
#define A_S_F       0.013975424859373686f /* 1/sqrt(64*64+32*32) */
#define A_V_F       0.015625f             /* 1/sqrt(64*32+32*64) = 1/64 */
#define INV_NN      0.25f                 /* 1/sqrt(16) */
#define EPS_F       1e-6f

__device__ __forceinline__ float sigm(float x) { return 1.0f / (1.0f + expf(-x)); }
__device__ __forceinline__ float silu_f(float x) { return x * sigm(x); }

#define WMMA16(A, B, C) \
  __builtin_amdgcn_wmma_f32_16x16x32_f16(false, (A), false, (B), (short)0, (C), false, false)

/* ---------------- utility ---------------- */
__global__ void k_zero(float* __restrict__ p, int n) {
  int i = blockIdx.x * blockDim.x + threadIdx.x;
  if (i < n) p[i] = 0.0f;
}

/* ---------------- weight packing into WMMA B-fragment order ----------------
   dst flat index = ((kc*NT + nt)*32 + lane)*16 + t
   true K = 32*kc + (t<8 ? t : t+8) + (lane<16 ? 0 : 8); (i,j) = (K/Jdim, K%Jdim); n = nt*16 + (lane&15)
   src value = W[i][j][n] * scale, W flat (i*Jdim+j)*Ncols+n                          */
__global__ void k_pack(_Float16* __restrict__ dst, const float* __restrict__ src,
                       int Jdim, int Ncols, int NT, int total, float scale) {
  int tid = blockIdx.x * blockDim.x + threadIdx.x;
  if (tid >= total) return;
  int t    = tid & 15;
  int lane = (tid >> 4) & 31;
  int rest = tid >> 9;
  int nt   = rest % NT;
  int kc   = rest / NT;
  int kg = 32 * kc + ((t < 8) ? t : (t + 8)) + ((lane < 16) ? 0 : 8);
  int i = kg / Jdim, j = kg % Jdim;
  int n = nt * 16 + (lane & 15);
  dst[tid] = (_Float16)(src[(i * Jdim + j) * Ncols + n] * scale);
}

/* ---------------- edge geometry + bessel basis ---------------- */
__global__ void k_geom(const float* __restrict__ pos, const int* __restrict__ esrc,
                       const int* __restrict__ edst, float* __restrict__ ea_s,
                       float* __restrict__ ea_v, float* __restrict__ basis) {
  int e = blockIdx.x * blockDim.x + threadIdx.x;
  if (e >= N_EDGES) return;
  int si = esrc[e], di = edst[e];
  float vx = pos[si * 3 + 0] - pos[di * 3 + 0];
  float vy = pos[si * 3 + 1] - pos[di * 3 + 1];
  float vz = pos[si * 3 + 2] - pos[di * 3 + 2];
  float r = sqrtf(vx * vx + vy * vy + vz * vz + 1e-12f);
  float rs = fmaxf(r, EPS_F);
  float inv = 1.0f / rs;
  float u = 2.0f * (r / 3.5f - 1.0f);
  float cut = 0.5f * (1.0f - cosf(PI_F * u));
  cut = (u > 0.0f) ? 0.0f : ((u < -1.0f) ? 1.0f : cut);
  ea_s[e] = cut;
  float c2 = cut * SQRT3_F * inv;
  ea_v[e * 3 + 0] = c2 * vx;
  ea_v[e * 3 + 1] = c2 * vy;
  ea_v[e * 3 + 2] = c2 * vz;
  float pref = sqrtf(2.0f / 3.5f) * sqrtf(8.0f);
  for (int b = 0; b < 8; ++b)
    basis[e * 8 + b] = pref * sinf((float)(b + 1) * PI_F * rs / 3.5f) * inv;
}

/* ---------------- embed + tp_self[0] ---------------- */
__global__ void k_embed(const int* __restrict__ atom_type, const float* __restrict__ embed,
                        const float* __restrict__ w_tp0, float* __restrict__ x8) {
  int n = blockIdx.x * blockDim.x + threadIdx.x;
  if (n >= N_NODES) return;
  int a = atom_type[n];
  float e8[8], o8[8];
  for (int i = 0; i < 8; ++i) { e8[i] = embed[a * 8 + i]; o8[i] = 0.0f; }
  for (int i = 0; i < 8; ++i)
    for (int j = 0; j < 8; ++j) {
      float p = e8[i] * e8[j];
      const float* wp = w_tp0 + (i * 8 + j) * 8;
      for (int k = 0; k < 8; ++k) o8[k] += p * wp[k];
    }
  for (int k = 0; k < 8; ++k) x8[n * 8 + k] = o8[k] * 0.125f;
}

/* ---------------- layer 0: node pre (sc + lin1) ---------------- */
__global__ void k_l0_pre(const float* __restrict__ x8, const float* __restrict__ sc0,
                         const float* __restrict__ lin1_0, float* __restrict__ scS,
                         float* __restrict__ xl) {
  int n = blockIdx.x * blockDim.x + threadIdx.x;
  if (n >= N_NODES) return;
  float xr[8];
  for (int i = 0; i < 8; ++i) xr[i] = x8[n * 8 + i];
  for (int k = 0; k < 96; ++k) {
    float a = 0.0f;
    for (int i = 0; i < 8; ++i) a += xr[i] * sc0[i * 96 + k];
    scS[n * 96 + k] = a * INV_SQRT8;
  }
  for (int u = 0; u < 8; ++u) {
    float a = 0.0f;
    for (int i = 0; i < 8; ++i) a += xr[i] * lin1_0[i * 8 + u];
    xl[n * 8 + u] = a * INV_SQRT8;
  }
}

/* ---------------- layer 0: edges, radial via WMMA; one wave = 16 edges ---- */
__global__ __launch_bounds__(32) void k_l0_edge_w(
    const float* __restrict__ basis, const float* __restrict__ ea_s,
    const float* __restrict__ ea_v, const int* __restrict__ esrc,
    const int* __restrict__ edst, const float* __restrict__ xl,
    const float* __restrict__ fw1, const _Float16* __restrict__ pw2,
    float* __restrict__ aggS, float* __restrict__ aggV) {
  __shared__ float h_lds[16][65];
  __shared__ float w_lds[16][17];
  const int lane = threadIdx.x;
  const int e0 = blockIdx.x * 16;
  /* phase 1: h = silu(basis @ fw1 / sqrt8), [16,64] */
  for (int idx = lane; idx < 16 * 64; idx += 32) {
    int m = idx >> 6, t = idx & 63;
    const float* bp = basis + (size_t)(e0 + m) * 8;
    float a = 0.0f;
    for (int q = 0; q < 8; ++q) a += bp[q] * fw1[q * 64 + t];
    h_lds[m][t] = silu_f(a * INV_SQRT8);
  }
  __syncthreads();
  /* phase 2: w16[16,16] = h @ (fw2/8) via 2 WMMAs */
  const int mA = lane & 15;
  const int kOffA = (lane & 16) ? 8 : 0;
  v16h a0, a1;
#pragma unroll
  for (int t = 0; t < 16; ++t) {
    const int kl = ((t < 8) ? t : (t + 8)) + kOffA;
    a0[t] = (_Float16)h_lds[mA][kl];
    a1[t] = (_Float16)h_lds[mA][32 + kl];
  }
  v8f acc = {0, 0, 0, 0, 0, 0, 0, 0};
  acc = WMMA16(a0, *(const v16h*)(pw2 + (size_t)lane * 16), acc);
  acc = WMMA16(a1, *(const v16h*)(pw2 + (size_t)(32 + lane) * 16), acc);
  const int nD = lane & 15;
  const int mOffD = (lane & 16) ? 8 : 0;
#pragma unroll
  for (int r = 0; r < 8; ++r) w_lds[r + mOffD][nD] = acc[r];
  __syncthreads();
  /* phase 3: messages + scatter, 2 lanes per edge */
  const int m = lane & 15;
  const int half = lane >> 4;
  const int e = e0 + m;
  const int si = esrc[e], di = edst[e];
  const float es = ea_s[e];
  const float ev0 = ea_v[e * 3], ev1 = ea_v[e * 3 + 1], ev2 = ea_v[e * 3 + 2];
  for (int uu = 0; uu < 4; ++uu) {
    int u = half * 4 + uu;
    float xe = xl[(size_t)si * 8 + u];
    atomicAdd(&aggS[(size_t)di * 8 + u], w_lds[m][u] * xe * es);
    float mv = w_lds[m][8 + u] * xe;
    atomicAdd(&aggV[(size_t)di * 24 + 0 * 8 + u], mv * ev0);
    atomicAdd(&aggV[(size_t)di * 24 + 1 * 8 + u], mv * ev1);
    atomicAdd(&aggV[(size_t)di * 24 + 2 * 8 + u], mv * ev2);
  }
}

/* ---------------- layer 0: node post (lin2 + tvnorm + gate) ---------------- */
__global__ void k_l0_post(const float* __restrict__ scS, const float* __restrict__ aggS,
                          const float* __restrict__ aggV, const float* __restrict__ l20s,
                          const float* __restrict__ l20v, float* __restrict__ s,
                          float* __restrict__ v) {
  int n = blockIdx.x * blockDim.x + threadIdx.x;
  if (n >= N_NODES) return;
  float as[8];
  for (int u = 0; u < 8; ++u) as[u] = aggS[n * 8 + u] * INV_NN;
  float ys[96];
  for (int k = 0; k < 96; ++k) {
    float a = 0.0f;
    for (int u = 0; u < 8; ++u) a += as[u] * l20s[u * 96 + k];
    ys[k] = C_S_F * scS[n * 96 + k] + C_X_F * INV_SQRT8 * a;
  }
  float ms = 0.0f;
  for (int k = 0; k < 96; ++k) ms += ys[k] * ys[k];
  float inv_s = rsqrtf(ms * (1.0f / 96.0f) + EPS_F);
  float yv[96];
  for (int c = 0; c < 3; ++c) {
    float av[8];
    for (int u = 0; u < 8; ++u) av[u] = aggV[n * 24 + c * 8 + u] * INV_NN;
    for (int k = 0; k < 32; ++k) {
      float a = 0.0f;
      for (int u = 0; u < 8; ++u) a += av[u] * l20v[u * 32 + k];
      yv[c * 32 + k] = a * INV_SQRT8;
    }
  }
  float vn = 0.0f;
  for (int k = 0; k < 32; ++k)
    vn += yv[k] * yv[k] + yv[32 + k] * yv[32 + k] + yv[64 + k] * yv[64 + k];
  float inv_v = rsqrtf(vn * (1.0f / 32.0f) + EPS_F);
  for (int k = 0; k < 64; ++k) s[n * 64 + k] = silu_f(ys[k] * inv_s);
  for (int k = 0; k < 32; ++k) {
    float g = sigm(ys[64 + k] * inv_s);
    v[n * 96 + 0 * 32 + k] = g * yv[k] * inv_v;
    v[n * 96 + 1 * 32 + k] = g * yv[32 + k] * inv_v;
    v[n * 96 + 2 * 32 + k] = g * yv[64 + k] * inv_v;
  }
}

/* ---------------- layers 1/2: node pre ---------------- */
__global__ void k_lay_pre(const float* __restrict__ s, const float* __restrict__ v,
                          const float* __restrict__ sc_s, const float* __restrict__ sc_v,
                          const float* __restrict__ l1s, const float* __restrict__ l1v,
                          float* __restrict__ scS, float* __restrict__ scV,
                          float* __restrict__ ls, float* __restrict__ lv) {
  int n = blockIdx.x * blockDim.x + threadIdx.x;
  if (n >= N_NODES) return;
  float sr[64];
  for (int u = 0; u < 64; ++u) sr[u] = s[(size_t)n * 64 + u];
  for (int k = 0; k < 96; ++k) {
    float a = 0.0f;
    for (int u = 0; u < 64; ++u) a += sr[u] * sc_s[u * 96 + k];
    scS[(size_t)n * 96 + k] = a * 0.125f;
  }
  for (int k = 0; k < 64; ++k) {
    float a = 0.0f;
    for (int u = 0; u < 64; ++u) a += sr[u] * l1s[u * 64 + k];
    ls[(size_t)n * 64 + k] = a * 0.125f;
  }
  float vr[96];
  for (int u = 0; u < 96; ++u) vr[u] = v[(size_t)n * 96 + u];
  for (int c = 0; c < 3; ++c)
    for (int k = 0; k < 32; ++k) {
      float a = 0.0f, b = 0.0f;
      for (int u = 0; u < 32; ++u) {
        float vv = vr[c * 32 + u];
        a += vv * sc_v[u * 32 + k];
        b += vv * l1v[u * 32 + k];
      }
      scV[(size_t)n * 96 + c * 32 + k] = a * INV_SQRT32;
      lv[(size_t)n * 96 + c * 32 + k] = b * INV_SQRT32;
    }
}

/* ---------------- layers 1/2: edges, radial via WMMA; one wave = 16 edges - */
__global__ __launch_bounds__(32) void k_lay_edge_w(
    const float* __restrict__ basis, const float* __restrict__ ea_s,
    const float* __restrict__ ea_v, const int* __restrict__ esrc,
    const int* __restrict__ edst, const float* __restrict__ ls,
    const float* __restrict__ lv, const float* __restrict__ fw1,
    const _Float16* __restrict__ pw2, float* __restrict__ aggS,
    float* __restrict__ aggV) {
  __shared__ float h_lds[16][65];
  __shared__ float w_lds[16][200];
  const int lane = threadIdx.x;
  const int e0 = blockIdx.x * 16;
  /* phase 1: h = silu(basis @ fw1 / sqrt8), [16,64] */
  for (int idx = lane; idx < 16 * 64; idx += 32) {
    int m = idx >> 6, t = idx & 63;
    const float* bp = basis + (size_t)(e0 + m) * 8;
    float a = 0.0f;
    for (int q = 0; q < 8; ++q) a += bp[q] * fw1[q * 64 + t];
    h_lds[m][t] = silu_f(a * INV_SQRT8);
  }
  __syncthreads();
  /* phase 2: w[16,192] = h @ (fw2/8): 2 K-chunks x 12 N-tiles of WMMA */
  const int mA = lane & 15;
  const int kOffA = (lane & 16) ? 8 : 0;
  v16h a0, a1;
#pragma unroll
  for (int t = 0; t < 16; ++t) {
    const int kl = ((t < 8) ? t : (t + 8)) + kOffA;
    a0[t] = (_Float16)h_lds[mA][kl];
    a1[t] = (_Float16)h_lds[mA][32 + kl];
  }
  const int nD = lane & 15;
  const int mOffD = (lane & 16) ? 8 : 0;
  for (int nt = 0; nt < 12; ++nt) {
    v8f acc = {0, 0, 0, 0, 0, 0, 0, 0};
    const _Float16* bp = pw2 + ((size_t)nt * 32 + lane) * 16;
    acc = WMMA16(a0, *(const v16h*)(bp), acc);
    acc = WMMA16(a1, *(const v16h*)(bp + 12 * 512), acc); /* kc=1 block */
#pragma unroll
    for (int r = 0; r < 8; ++r) w_lds[r + mOffD][nt * 16 + nD] = acc[r];
  }
  __syncthreads();
  /* phase 3: messages + scatter, 2 lanes per edge */
  const int m = lane & 15;
  const int half = lane >> 4;
  const int e = e0 + m;
  const int si = esrc[e], di = edst[e];
  const float es = ea_s[e];
  const float ev0 = ea_v[e * 3], ev1 = ea_v[e * 3 + 1], ev2 = ea_v[e * 3 + 2];
  const float* lsp = ls + (size_t)si * 64;
  const float* lvp = lv + (size_t)si * 96;
  float* aS = aggS + (size_t)di * 96;
  float* aV = aggV + (size_t)di * 288;
  const float* wrow = w_lds[m];
  for (int uu = 0; uu < 32; ++uu) { /* m0: 0e x 0e -> 0e */
    int u = half * 32 + uu;
    atomicAdd(aS + u, wrow[u] * lsp[u] * es);
  }
  for (int uu = 0; uu < 32; ++uu) { /* m1: 0e x 1o -> 1o */
    int u = half * 32 + uu;
    float base = wrow[64 + u] * lsp[u];
    atomicAdd(aV + 0 * 96 + u, base * ev0);
    atomicAdd(aV + 1 * 96 + u, base * ev1);
    atomicAdd(aV + 2 * 96 + u, base * ev2);
  }
  for (int uu = 0; uu < 16; ++uu) { /* m2: 1o x 0e -> 1o */
    int u = half * 16 + uu;
    float wv = wrow[128 + u] * es;
    atomicAdd(aV + 0 * 96 + 64 + u, wv * lvp[0 * 32 + u]);
    atomicAdd(aV + 1 * 96 + 64 + u, wv * lvp[1 * 32 + u]);
    atomicAdd(aV + 2 * 96 + 64 + u, wv * lvp[2 * 32 + u]);
  }
  for (int uu = 0; uu < 16; ++uu) { /* m3: 1o x 1o -> 0e */
    int u = half * 16 + uu;
    float dv = lvp[u] * ev0 + lvp[32 + u] * ev1 + lvp[64 + u] * ev2;
    atomicAdd(aS + 64 + u, wrow[160 + u] * dv * INV_SQRT3_F);
  }
}

/* ---------------- layers 1/2: node post (lin2 + tvnorm + gate) ---------------- */
__global__ void k_lay_post(const float* __restrict__ scS, const float* __restrict__ scV,
                           const float* __restrict__ aggS, const float* __restrict__ aggV,
                           const float* __restrict__ l2s, const float* __restrict__ l2v,
                           float* __restrict__ gs, float* __restrict__ gv) {
  int n = blockIdx.x * blockDim.x + threadIdx.x;
  if (n >= N_NODES) return;
  float ag[96];
  for (int u = 0; u < 96; ++u) ag[u] = aggS[(size_t)n * 96 + u] * INV_NN;
  float ys[96];
  for (int k = 0; k < 96; ++k) {
    float a = 0.0f;
    for (int u = 0; u < 96; ++u) a += ag[u] * l2s[u * 96 + k];
    ys[k] = C_S_F * scS[(size_t)n * 96 + k] + C_X_F * INV_SQRT96 * a;
  }
  float ms = 0.0f;
  for (int k = 0; k < 96; ++k) ms += ys[k] * ys[k];
  float inv_s = rsqrtf(ms * (1.0f / 96.0f) + EPS_F);
  float yv[96];
  for (int c = 0; c < 3; ++c) {
    float av[96];
    for (int u = 0; u < 96; ++u) av[u] = aggV[(size_t)n * 288 + c * 96 + u] * INV_NN;
    for (int k = 0; k < 32; ++k) {
      float a = 0.0f;
      for (int u = 0; u < 96; ++u) a += av[u] * l2v[u * 32 + k];
      yv[c * 32 + k] = C_S_F * scV[(size_t)n * 96 + c * 32 + k] + C_X_F * INV_SQRT96 * a;
    }
  }
  float vn = 0.0f;
  for (int k = 0; k < 32; ++k)
    vn += yv[k] * yv[k] + yv[32 + k] * yv[32 + k] + yv[64 + k] * yv[64 + k];
  float inv_v = rsqrtf(vn * (1.0f / 32.0f) + EPS_F);
  for (int k = 0; k < 64; ++k) gs[(size_t)n * 64 + k] = silu_f(ys[k] * inv_s);
  for (int k = 0; k < 32; ++k) {
    float g = sigm(ys[64 + k] * inv_s);
    gv[(size_t)n * 96 + 0 * 32 + k] = g * yv[k] * inv_v;
    gv[(size_t)n * 96 + 1 * 32 + k] = g * yv[32 + k] * inv_v;
    gv[(size_t)n * 96 + 2 * 32 + k] = g * yv[64 + k] * inv_v;
  }
}

/* ---------------- tp_hidden via WMMA: one wave = 16 nodes ---------------- */
__global__ __launch_bounds__(32) void k_tp_hidden(
    float* __restrict__ s, float* __restrict__ v, const float* __restrict__ gs,
    const float* __restrict__ gv, const _Float16* __restrict__ pss,
    const _Float16* __restrict__ pvv, const _Float16* __restrict__ psv,
    const _Float16* __restrict__ pvs) {
  __shared__ float s_lds[16][65];
  __shared__ float v_lds[16][3][33];
  const int lane = threadIdx.x;
  const int node0 = blockIdx.x * 16;
  for (int idx = lane; idx < 16 * 64; idx += 32)
    s_lds[idx >> 6][idx & 63] = gs[(size_t)node0 * 64 + idx];
  for (int idx = lane; idx < 16 * 96; idx += 32) {
    int m = idx / 96, q = idx - m * 96;
    v_lds[m][q >> 5][q & 31] = gv[(size_t)node0 * 96 + idx];
  }
  __syncthreads();
  const int mA = lane & 15;
  const int kOffA = (lane & 16) ? 8 : 0;
  const int nD = lane & 15;
  const int mOffD = (lane & 16) ? 8 : 0;

  /* ---- scalar output D[16,64] = P_ss @ Wss + P_vv @ (Wvv/sqrt3) ---- */
  v8f acc0 = {0, 0, 0, 0, 0, 0, 0, 0};
  v8f acc1 = acc0, acc2 = acc0, acc3 = acc0;
  for (int kc = 0; kc < 128; ++kc) {
    const int i = kc >> 1, jb = (kc & 1) << 5;
    const float si = s_lds[mA][i];
    v16h a;
#pragma unroll
    for (int t = 0; t < 16; ++t) {
      const int kl = ((t < 8) ? t : (t + 8)) + kOffA;
      a[t] = (_Float16)(si * s_lds[mA][jb + kl]);
    }
    const _Float16* bp = pss + ((size_t)(kc * 4) * 32 + lane) * 16;
    __builtin_prefetch(bp + 2048, 0, 1); /* next kc block -> global_prefetch_b8 */
    acc0 = WMMA16(a, *(const v16h*)(bp), acc0);
    acc1 = WMMA16(a, *(const v16h*)(bp + 512), acc1);
    acc2 = WMMA16(a, *(const v16h*)(bp + 1024), acc2);
    acc3 = WMMA16(a, *(const v16h*)(bp + 1536), acc3);
  }
  for (int kc = 0; kc < 32; ++kc) {
    const float vi0 = v_lds[mA][0][kc], vi1 = v_lds[mA][1][kc], vi2 = v_lds[mA][2][kc];
    v16h a;
#pragma unroll
    for (int t = 0; t < 16; ++t) {
      const int kl = ((t < 8) ? t : (t + 8)) + kOffA;
      a[t] = (_Float16)(vi0 * v_lds[mA][0][kl] + vi1 * v_lds[mA][1][kl] +
                        vi2 * v_lds[mA][2][kl]);
    }
    const _Float16* bp = pvv + ((size_t)(kc * 4) * 32 + lane) * 16;
    acc0 = WMMA16(a, *(const v16h*)(bp), acc0);
    acc1 = WMMA16(a, *(const v16h*)(bp + 512), acc1);
    acc2 = WMMA16(a, *(const v16h*)(bp + 1024), acc2);
    acc3 = WMMA16(a, *(const v16h*)(bp + 1536), acc3);
  }
#pragma unroll
  for (int r = 0; r < 8; ++r) {
    const int m = r + mOffD;
    float* sp = s + (size_t)(node0 + m) * 64 + nD;
    sp[0]  += 0.1f * A_S_F * acc0[r];
    sp[16] += 0.1f * A_S_F * acc1[r];
    sp[32] += 0.1f * A_S_F * acc2[r];
    sp[48] += 0.1f * A_S_F * acc3[r];
  }

  /* ---- vector output per c: D[16,32] = (s x v_c)@Wsv + (v_c x s)@Wvs ---- */
  for (int c = 0; c < 3; ++c) {
    v8f va0 = {0, 0, 0, 0, 0, 0, 0, 0};
    v8f va1 = va0;
    for (int kc = 0; kc < 64; ++kc) { /* sv: k=i*32+j, i=kc */
      const float si = s_lds[mA][kc];
      v16h a;
#pragma unroll
      for (int t = 0; t < 16; ++t) {
        const int kl = ((t < 8) ? t : (t + 8)) + kOffA;
        a[t] = (_Float16)(si * v_lds[mA][c][kl]);
      }
      const _Float16* bp = psv + ((size_t)(kc * 2) * 32 + lane) * 16;
      va0 = WMMA16(a, *(const v16h*)(bp), va0);
      va1 = WMMA16(a, *(const v16h*)(bp + 512), va1);
    }
    for (int kc = 0; kc < 64; ++kc) { /* vs: k=i*64+j, i=kc>>1 */
      const int i = kc >> 1, jb = (kc & 1) << 5;
      const float vi = v_lds[mA][c][i];
      v16h a;
#pragma unroll
      for (int t = 0; t < 16; ++t) {
        const int kl = ((t < 8) ? t : (t + 8)) + kOffA;
        a[t] = (_Float16)(vi * s_lds[mA][jb + kl]);
      }
      const _Float16* bp = pvs + ((size_t)(kc * 2) * 32 + lane) * 16;
      va0 = WMMA16(a, *(const v16h*)(bp), va0);
      va1 = WMMA16(a, *(const v16h*)(bp + 512), va1);
    }
#pragma unroll
    for (int r = 0; r < 8; ++r) {
      const int m = r + mOffD;
      float* vp = v + (size_t)(node0 + m) * 96 + c * 32 + nD;
      vp[0]  += 0.1f * A_V_F * va0[r];
      vp[16] += 0.1f * A_V_F * va1[r];
    }
  }
}

/* ---------------- penultimate WMMA + silu + quadratic readout ---------------- */
__global__ __launch_bounds__(32) void k_pen(
    const float* __restrict__ s, const float* __restrict__ v,
    const _Float16* __restrict__ pss, const _Float16* __restrict__ pvv,
    const float* __restrict__ w_last, const int* __restrict__ batch,
    float* __restrict__ out) {
  __shared__ float s_lds[16][65];
  __shared__ float v_lds[16][3][33];
  __shared__ float x_lds[16][17];
  const int lane = threadIdx.x;
  const int node0 = blockIdx.x * 16;
  for (int idx = lane; idx < 16 * 64; idx += 32)
    s_lds[idx >> 6][idx & 63] = s[(size_t)node0 * 64 + idx];
  for (int idx = lane; idx < 16 * 96; idx += 32) {
    int m = idx / 96, q = idx - m * 96;
    v_lds[m][q >> 5][q & 31] = v[(size_t)node0 * 96 + idx];
  }
  __syncthreads();
  const int mA = lane & 15;
  const int kOffA = (lane & 16) ? 8 : 0;
  v8f acc = {0, 0, 0, 0, 0, 0, 0, 0};
  for (int kc = 0; kc < 128; ++kc) {
    const int i = kc >> 1, jb = (kc & 1) << 5;
    const float si = s_lds[mA][i];
    v16h a;
#pragma unroll
    for (int t = 0; t < 16; ++t) {
      const int kl = ((t < 8) ? t : (t + 8)) + kOffA;
      a[t] = (_Float16)(si * s_lds[mA][jb + kl]);
    }
    acc = WMMA16(a, *(const v16h*)(pss + ((size_t)kc * 32 + lane) * 16), acc);
  }
  for (int kc = 0; kc < 32; ++kc) {
    const float vi0 = v_lds[mA][0][kc], vi1 = v_lds[mA][1][kc], vi2 = v_lds[mA][2][kc];
    v16h a;
#pragma unroll
    for (int t = 0; t < 16; ++t) {
      const int kl = ((t < 8) ? t : (t + 8)) + kOffA;
      a[t] = (_Float16)(vi0 * v_lds[mA][0][kl] + vi1 * v_lds[mA][1][kl] +
                        vi2 * v_lds[mA][2][kl]);
    }
    acc = WMMA16(a, *(const v16h*)(pvv + ((size_t)kc * 32 + lane) * 16), acc);
  }
  const int nD = lane & 15;
  const int mOffD = (lane & 16) ? 8 : 0;
#pragma unroll
  for (int r = 0; r < 8; ++r) {
    float z = A_S_F * acc[r]; /* a_pen == a_s */
    x_lds[r + mOffD][nD] = silu_f(z);
  }
  __syncthreads();
  if (lane < 16) {
    float xr[16];
    for (int j = 0; j < 16; ++j) xr[j] = x_lds[lane][j];
    float a2 = 0.0f;
    for (int i = 0; i < 16; ++i)
      for (int j = 0; j < 16; ++j) a2 += xr[i] * xr[j] * w_last[i * 16 + j];
    atomicAdd(&out[batch[node0 + lane]], a2 * (1.0f / 16.0f) * 0.05f);
  }
}

/* ================================ launcher ================================ */
extern "C" void kernel_launch(void* const* d_in, const int* in_sizes, int n_in,
                              void* d_out, int out_size, void* d_ws, size_t ws_size,
                              hipStream_t stream) {
  (void)in_sizes; (void)n_in; (void)out_size; (void)ws_size;
  const float* pos    = (const float*)d_in[0];
  const int*   atomt  = (const int*)d_in[1];
  const int*   esrc   = (const int*)d_in[2];
  const int*   edst   = (const int*)d_in[3];
  const int*   batch  = (const int*)d_in[4];
  const float* embed  = (const float*)d_in[5];
  const float* w_tp0  = (const float*)d_in[6];
  const float* tpss   = (const float*)d_in[7];
  const float* tpvv   = (const float*)d_in[8];
  const float* tpsv   = (const float*)d_in[9];
  const float* tpvs   = (const float*)d_in[10];
  const float* penss  = (const float*)d_in[11];
  const float* penvv  = (const float*)d_in[12];
  const float* w_last = (const float*)d_in[13];
  const float* sc0    = (const float*)d_in[14];
  const float* lin1_0 = (const float*)d_in[15];
  const float* fc0w1  = (const float*)d_in[16];
  const float* fc0w2  = (const float*)d_in[17];
  const float* l20s   = (const float*)d_in[18];
  const float* l20v   = (const float*)d_in[19];
  const float* sc_s   = (const float*)d_in[20];
  const float* sc_v   = (const float*)d_in[21];
  const float* l1s    = (const float*)d_in[22];
  const float* l1v    = (const float*)d_in[23];
  const float* fcw1   = (const float*)d_in[24];
  const float* fcw2   = (const float*)d_in[25];
  const float* l2s    = (const float*)d_in[26];
  const float* l2v    = (const float*)d_in[27];
  float* out = (float*)d_out;

  float* w = (float*)d_ws;
  size_t o = 0;
  float* ea_s = w + o; o += (size_t)N_EDGES;
  float* ea_v = w + o; o += (size_t)3 * N_EDGES;
  float* basis= w + o; o += (size_t)8 * N_EDGES;
  float* x8   = w + o; o += (size_t)8 * N_NODES;
  float* scS  = w + o; o += (size_t)96 * N_NODES;
  float* scV  = w + o; o += (size_t)96 * N_NODES;
  float* ls   = w + o; o += (size_t)64 * N_NODES;
  float* lv   = w + o; o += (size_t)96 * N_NODES;
  float* aggS = w + o; o += (size_t)96 * N_NODES;
  float* aggV = w + o; o += (size_t)288 * N_NODES;
  float* sbuf = w + o; o += (size_t)64 * N_NODES;
  float* vbuf = w + o; o += (size_t)96 * N_NODES;
  float* gsb  = w + o; o += (size_t)64 * N_NODES;
  float* gvb  = w + o; o += (size_t)96 * N_NODES;
  _Float16* hreg = (_Float16*)(w + o);
  size_t ho = 0;
  _Float16* pk_ss[2]; _Float16* pk_vv[2]; _Float16* pk_sv[2]; _Float16* pk_vs[2];
  _Float16* pk_rw2[2];
  for (int i = 0; i < 2; ++i) {
    pk_ss[i]  = hreg + ho; ho += 262144;
    pk_vv[i]  = hreg + ho; ho += 65536;
    pk_sv[i]  = hreg + ho; ho += 65536;
    pk_vs[i]  = hreg + ho; ho += 65536;
    pk_rw2[i] = hreg + ho; ho += 12288;
  }
  _Float16* pk_pss = hreg + ho; ho += 65536;
  _Float16* pk_pvv = hreg + ho; ho += 16384;
  _Float16* pk_r0  = hreg + ho; ho += 1024;

  const int TB = 256;
  const int gE = (N_EDGES + TB - 1) / TB;
  const int gN = (N_NODES + TB - 1) / TB;
  const int gTile = N_NODES / 16;  /* 1250 */
  const int gETile = N_EDGES / 16; /* 20000 */

  /* output zeroing (harness poisons d_out) */
  k_zero<<<1, 64, 0, stream>>>(out, N_GRAPHS);

  /* pack WMMA B-fragments from f32 weights */
  for (int i = 0; i < 2; ++i) {
    k_pack<<<(262144 + TB - 1) / TB, TB, 0, stream>>>(pk_ss[i], tpss + (size_t)i * 262144, 64, 64, 4, 262144, 1.0f);
    k_pack<<<(65536 + TB - 1) / TB, TB, 0, stream>>>(pk_vv[i], tpvv + (size_t)i * 65536, 32, 64, 4, 65536, INV_SQRT3_F);
    k_pack<<<(65536 + TB - 1) / TB, TB, 0, stream>>>(pk_sv[i], tpsv + (size_t)i * 65536, 32, 32, 2, 65536, 1.0f);
    k_pack<<<(65536 + TB - 1) / TB, TB, 0, stream>>>(pk_vs[i], tpvs + (size_t)i * 65536, 64, 32, 2, 65536, 1.0f);
    k_pack<<<(12288 + TB - 1) / TB, TB, 0, stream>>>(pk_rw2[i], fcw2 + (size_t)i * 12288, 64, 192, 12, 12288, 0.125f);
  }
  k_pack<<<(65536 + TB - 1) / TB, TB, 0, stream>>>(pk_pss, penss, 64, 16, 1, 65536, 1.0f);
  k_pack<<<(16384 + TB - 1) / TB, TB, 0, stream>>>(pk_pvv, penvv, 32, 16, 1, 16384, INV_SQRT3_F);
  k_pack<<<(1024 + TB - 1) / TB, TB, 0, stream>>>(pk_r0, fc0w2, 64, 16, 1, 1024, 0.125f);

  /* geometry + embedding */
  k_geom<<<gE, TB, 0, stream>>>(pos, esrc, edst, ea_s, ea_v, basis);
  k_embed<<<gN, TB, 0, stream>>>(atomt, embed, w_tp0, x8);

  /* layer 0 */
  k_l0_pre<<<gN, TB, 0, stream>>>(x8, sc0, lin1_0, scS, ls);
  k_zero<<<(8 * N_NODES + TB - 1) / TB, TB, 0, stream>>>(aggS, 8 * N_NODES);
  k_zero<<<(24 * N_NODES + TB - 1) / TB, TB, 0, stream>>>(aggV, 24 * N_NODES);
  k_l0_edge_w<<<gETile, 32, 0, stream>>>(basis, ea_s, ea_v, esrc, edst, ls, fc0w1,
                                         pk_r0, aggS, aggV);
  k_l0_post<<<gN, TB, 0, stream>>>(scS, aggS, aggV, l20s, l20v, sbuf, vbuf);

  /* layers 1 and 2 */
  for (int i = 0; i < 2; ++i) {
    k_lay_pre<<<gN, TB, 0, stream>>>(sbuf, vbuf, sc_s + (size_t)i * 6144,
                                     sc_v + (size_t)i * 1024, l1s + (size_t)i * 4096,
                                     l1v + (size_t)i * 1024, scS, scV, ls, lv);
    k_zero<<<(96 * N_NODES + TB - 1) / TB, TB, 0, stream>>>(aggS, 96 * N_NODES);
    k_zero<<<(288 * N_NODES + TB - 1) / TB, TB, 0, stream>>>(aggV, 288 * N_NODES);
    k_lay_edge_w<<<gETile, 32, 0, stream>>>(basis, ea_s, ea_v, esrc, edst, ls, lv,
                                            fcw1 + (size_t)i * 512, pk_rw2[i],
                                            aggS, aggV);
    k_lay_post<<<gN, TB, 0, stream>>>(scS, scV, aggS, aggV, l2s + (size_t)i * 9216,
                                      l2v + (size_t)i * 3072, gsb, gvb);
    k_tp_hidden<<<gTile, 32, 0, stream>>>(sbuf, vbuf, gsb, gvb, pk_ss[i], pk_vv[i],
                                          pk_sv[i], pk_vs[i]);
  }

  /* penultimate + readout */
  k_pen<<<gTile, 32, 0, stream>>>(sbuf, vbuf, pk_pss, pk_pvv, w_last, batch, out);
}
